// customGraphSAGE2_13984413516241
// MI455X (gfx1250) — compile-verified, matrix operand order checked by reading, not run on hardware
//
#include <hip/hip_runtime.h>

typedef __bf16 bf16;
typedef __attribute__((ext_vector_type(16))) __bf16 v16bf;
typedef __attribute__((ext_vector_type(8)))  float   v8f;

#define ND 128
#define ED 64

// ---------------- WMMA helpers ----------------

__device__ __forceinline__ v8f vzero8() {
  v8f z;
#pragma unroll
  for (int i = 0; i < 8; ++i) z[i] = 0.0f;
  return z;
}

// A fragment (16x32 bf16) from LDS row-major tile. row points at lds[m][0].
// Per ISA 7.12.2: lanes 0-15 hold K = kb+[0..7],[16..23]; lanes 16-31 K+8.
__device__ __forceinline__ v16bf ldsA(const bf16* row, int kb, int g) {
  union { v16bf v; float4 f[2]; } u;
  const bf16* p = row + kb + 8 * g;
  u.f[0] = *(const float4*)(p);
  u.f[1] = *(const float4*)(p + 16);
  return u.v;
}

// B fragment from pre-packed weights: fragment f stores lane-major 32B runs.
__device__ __forceinline__ v16bf ldB(const bf16* p) {
  union { v16bf v; float4 f[2]; } u;
  u.f[0] = ((const float4*)p)[0];
  u.f[1] = ((const float4*)p)[1];
  return u.v;
}

__device__ __forceinline__ v8f wmma_bf16(v16bf a, v16bf b, v8f c) {
  // (neg_a, A, neg_b, B, c_mod, C, reuse_a, reuse_b)
  return __builtin_amdgcn_wmma_f32_16x16x32_bf16(false, a, false, b, (short)0, c,
                                                 false, false);
}

// ---------------- weight packing ----------------
// Pack W[K][N] (f32 row-major) into bf16 WMMA B fragments.
// frag f = (kb/32)*(N/16) + (nb/16); within frag: lane L holds 16 bf16:
//   n = nb + (L&15); k = kb + 8*(L>>4) + (i<8 ? i : i+8)
__global__ __launch_bounds__(256) void pack_weight(const float* __restrict__ W,
                                                   bf16* __restrict__ dst,
                                                   int K, int N) {
  int total = K * N;
  int ntiles = N >> 4;
  for (int idx = blockIdx.x * blockDim.x + threadIdx.x; idx < total;
       idx += gridDim.x * blockDim.x) {
    int f = idx >> 9;
    int r = idx & 511;
    int L = r >> 4;
    int i = r & 15;
    int kb = (f / ntiles) << 5;
    int nb = (f % ntiles) << 4;
    int n  = nb + (L & 15);
    int k  = kb + 8 * (L >> 4) + ((i < 8) ? i : (i + 8));
    dst[idx] = (bf16)W[k * N + n];
  }
}

// ---------------- degree ----------------

__global__ __launch_bounds__(256) void deg_count(const int* __restrict__ dstIdx,
                                                 float* __restrict__ deg, int E) {
  int e = blockIdx.x * blockDim.x + threadIdx.x;
  if (e < E) atomicAdd(&deg[dstIdx[e]], 1.0f);
}

__global__ __launch_bounds__(256) void deg_recip(const float* __restrict__ deg,
                                                 float* __restrict__ recip, int N) {
  int i = blockIdx.x * blockDim.x + threadIdx.x;
  if (i < N) recip[i] = 1.0f / fmaxf(deg[i], 1.0f);
}

// ---------------- SAGE conv: edge message GEMM + scatter ----------------
// msg[e,:] = concat(x[src[e]], ef[e]) @ W_nb + b_nb ; atomic-add into agg[dst[e]]
__global__ __launch_bounds__(128) void sage_msg(
    const float* __restrict__ x, const int* __restrict__ src,
    const int* __restrict__ dstIdx, const float* __restrict__ ef,
    const bf16* __restrict__ Wnbp, const float* __restrict__ bnb,
    float* __restrict__ agg) {
  __shared__ __align__(16) bf16 sPair[16 * 192];
  __shared__ int sSrc[16], sDst[16];
  const int tid = threadIdx.x, w = tid >> 5, L = tid & 31, g = L >> 4, m0 = L & 15;
  const int e0 = blockIdx.x * 16;
  if (tid < 16) { sSrc[tid] = src[e0 + tid]; sDst[tid] = dstIdx[e0 + tid]; }
  __syncthreads();
  for (int idx = tid; idx < 16 * 192; idx += 128) {
    int m = idx / 192, k = idx - m * 192;
    float v = (k < 128) ? x[(size_t)sSrc[m] * ND + k]
                        : ef[(size_t)(e0 + m) * ED + (k - 128)];
    sPair[idx] = (bf16)v;
  }
  __syncthreads();
  v8f acc0 = vzero8(), acc1 = vzero8();
  const bf16* row = sPair + m0 * 192;
#pragma unroll
  for (int ks = 0; ks < 6; ++ks) {
    v16bf a = ldsA(row, ks * 32, g);
    const bf16* bb = Wnbp + (size_t)(ks * 8 + w * 2) * 512 + L * 16;  // N/16 = 8
    acc0 = wmma_bf16(a, ldB(bb), acc0);
    acc1 = wmma_bf16(a, ldB(bb + 512), acc1);
  }
  const int c0 = w * 32 + m0;
  const float bb0 = bnb[c0], bb1 = bnb[c0 + 16];
#pragma unroll
  for (int r = 0; r < 8; ++r) {
    int m = r + 8 * g;
    int node = sDst[m];
    atomicAdd(&agg[(size_t)node * ND + c0],      acc0[r] + bb0);
    atomicAdd(&agg[(size_t)node * ND + c0 + 16], acc1[r] + bb1);
  }
}

// ---------------- SAGE conv: node update ----------------
// xout = relu(xin @ W_self + b_self + agg * recip)
__global__ __launch_bounds__(128) void sage_node(
    const float* __restrict__ xin, const bf16* __restrict__ Wselfp,
    const float* __restrict__ bself, const float* __restrict__ agg,
    const float* __restrict__ recip, float* __restrict__ xout) {
  __shared__ __align__(16) bf16 sX[16 * 128];
  const int tid = threadIdx.x, w = tid >> 5, L = tid & 31, g = L >> 4, m0 = L & 15;
  const int n0 = blockIdx.x * 16;
  for (int idx = tid; idx < 16 * 128; idx += 128)
    sX[idx] = (bf16)xin[(size_t)n0 * ND + idx];
  __syncthreads();
  v8f acc0 = vzero8(), acc1 = vzero8();
  const bf16* row = sX + m0 * 128;
#pragma unroll
  for (int ks = 0; ks < 4; ++ks) {
    v16bf a = ldsA(row, ks * 32, g);
    const bf16* bb = Wselfp + (size_t)(ks * 8 + w * 2) * 512 + L * 16;  // N/16 = 8
    acc0 = wmma_bf16(a, ldB(bb), acc0);
    acc1 = wmma_bf16(a, ldB(bb + 512), acc1);
  }
  const int c0 = w * 32 + m0;
  const float b0 = bself[c0], b1 = bself[c0 + 16];
#pragma unroll
  for (int r = 0; r < 8; ++r) {
    int node = n0 + r + 8 * g;
    float rc = recip[node];
    float v0 = acc0[r] + b0 + agg[(size_t)node * ND + c0] * rc;
    float v1 = acc1[r] + b1 + agg[(size_t)node * ND + c0 + 16] * rc;
    xout[(size_t)node * ND + c0]      = fmaxf(v0, 0.f);
    xout[(size_t)node * ND + c0 + 16] = fmaxf(v1, 0.f);
  }
}

// ---------------- MLP pass 1: Z1 column stats ----------------
__global__ __launch_bounds__(256) void mlp_stats1(
    const float* __restrict__ x2, const int* __restrict__ vi,
    const int* __restrict__ ui, const float* __restrict__ ef,
    const bf16* __restrict__ W1p, const float* __restrict__ b1,
    float* __restrict__ sum1, float* __restrict__ sq1, int ntiles) {
  __shared__ __align__(16) bf16 sPair[16 * 320];
  __shared__ int sV[16], sU[16];
  const int tid = threadIdx.x, w = tid >> 5, L = tid & 31, g = L >> 4, m0 = L & 15;
  float b1c[4];
#pragma unroll
  for (int i = 0; i < 4; ++i) b1c[i] = b1[w * 64 + i * 16 + m0];
  float sumP[4] = {0, 0, 0, 0}, sqP[4] = {0, 0, 0, 0};
  for (int t = blockIdx.x; t < ntiles; t += gridDim.x) {
    const int e0 = t * 16;
    __syncthreads();
    if (tid < 16) { sV[tid] = vi[e0 + tid]; sU[tid] = ui[e0 + tid]; }
    __syncthreads();
    __builtin_prefetch(ef + (size_t)(e0 + 16) * ED, 0, 0);
    for (int idx = tid; idx < 16 * 320; idx += 256) {
      int m = idx / 320, k = idx - m * 320;
      float v;
      if (k < 128)      v = x2[(size_t)sV[m] * ND + k];
      else if (k < 256) v = x2[(size_t)sU[m] * ND + (k - 128)];
      else              v = ef[(size_t)(e0 + m) * ED + (k - 256)];
      sPair[idx] = (bf16)v;
    }
    __syncthreads();
    const bf16* row = sPair + m0 * 320;
    v8f acc[4];
#pragma unroll
    for (int i = 0; i < 4; ++i) acc[i] = vzero8();
#pragma unroll
    for (int ks = 0; ks < 10; ++ks) {
      v16bf a = ldsA(row, ks * 32, g);
      const bf16* bb = W1p + (size_t)(ks * 32 + w * 4) * 512 + L * 16;  // N/16 = 32
#pragma unroll
      for (int i = 0; i < 4; ++i) acc[i] = wmma_bf16(a, ldB(bb + i * 512), acc[i]);
    }
#pragma unroll
    for (int i = 0; i < 4; ++i) {
      float s = 0.f, q = 0.f;
#pragma unroll
      for (int r = 0; r < 8; ++r) { float z = acc[i][r] + b1c[i]; s += z; q += z * z; }
      sumP[i] += s; sqP[i] += q;
    }
  }
#pragma unroll
  for (int i = 0; i < 4; ++i) {
    float s = sumP[i] + __shfl_xor(sumP[i], 16, 32);
    float q = sqP[i]  + __shfl_xor(sqP[i], 16, 32);
    if (L < 16) {
      int col = w * 64 + i * 16 + L;
      atomicAdd(&sum1[col], s);
      atomicAdd(&sq1[col], q);
    }
  }
}

// ---------------- BN finalize: a = g*rsqrt(var+eps); c = beta - mean*a ------
__global__ __launch_bounds__(256) void bn_finalize(
    const float* __restrict__ sum, const float* __restrict__ sq,
    const float* __restrict__ gamma, const float* __restrict__ beta,
    float* __restrict__ a, float* __restrict__ c, int N, float invE) {
  int j = blockIdx.x * blockDim.x + threadIdx.x;
  if (j < N) {
    float m = sum[j] * invE;
    float v = fmaxf(sq[j] * invE - m * m, 0.0f);
    float s = gamma[j] * rsqrtf(v + 1e-5f);
    a[j] = s;
    c[j] = beta[j] - m * s;
  }
}

// ---------------- MLP pass 2: recompute Z1->h1, Z2 column stats ----------------
__global__ __launch_bounds__(256) void mlp_stats2(
    const float* __restrict__ x2, const int* __restrict__ vi,
    const int* __restrict__ ui, const float* __restrict__ ef,
    const bf16* __restrict__ W1p, const float* __restrict__ b1,
    const float* __restrict__ a1, const float* __restrict__ c1,
    const bf16* __restrict__ W2p, const float* __restrict__ b2,
    float* __restrict__ sum2, float* __restrict__ sq2, int ntiles) {
  __shared__ __align__(16) bf16 sPair[16 * 320];
  __shared__ __align__(16) bf16 sH1[16 * 512];
  __shared__ int sV[16], sU[16];
  const int tid = threadIdx.x, w = tid >> 5, L = tid & 31, g = L >> 4, m0 = L & 15;
  float aEff[4], cEff[4];
#pragma unroll
  for (int i = 0; i < 4; ++i) {
    int col = w * 64 + i * 16 + m0;
    float aa = a1[col];
    aEff[i] = aa;
    cEff[i] = b1[col] * aa + c1[col];
  }
  float b2c[2];
#pragma unroll
  for (int i = 0; i < 2; ++i) b2c[i] = b2[w * 32 + i * 16 + m0];
  float sumP[2] = {0, 0}, sqP[2] = {0, 0};
  for (int t = blockIdx.x; t < ntiles; t += gridDim.x) {
    const int e0 = t * 16;
    __syncthreads();
    if (tid < 16) { sV[tid] = vi[e0 + tid]; sU[tid] = ui[e0 + tid]; }
    __syncthreads();
    __builtin_prefetch(ef + (size_t)(e0 + 16) * ED, 0, 0);
    for (int idx = tid; idx < 16 * 320; idx += 256) {
      int m = idx / 320, k = idx - m * 320;
      float v;
      if (k < 128)      v = x2[(size_t)sV[m] * ND + k];
      else if (k < 256) v = x2[(size_t)sU[m] * ND + (k - 128)];
      else              v = ef[(size_t)(e0 + m) * ED + (k - 256)];
      sPair[idx] = (bf16)v;
    }
    __syncthreads();
    {  // GEMM1 + BN1 + relu -> sH1
      const bf16* row = sPair + m0 * 320;
      v8f acc[4];
#pragma unroll
      for (int i = 0; i < 4; ++i) acc[i] = vzero8();
#pragma unroll
      for (int ks = 0; ks < 10; ++ks) {
        v16bf a = ldsA(row, ks * 32, g);
        const bf16* bb = W1p + (size_t)(ks * 32 + w * 4) * 512 + L * 16;
#pragma unroll
        for (int i = 0; i < 4; ++i) acc[i] = wmma_bf16(a, ldB(bb + i * 512), acc[i]);
      }
#pragma unroll
      for (int i = 0; i < 4; ++i) {
        int col = w * 64 + i * 16 + m0;
#pragma unroll
        for (int r = 0; r < 8; ++r) {
          int m = r + 8 * g;
          sH1[m * 512 + col] = (bf16)fmaxf(acc[i][r] * aEff[i] + cEff[i], 0.0f);
        }
      }
    }
    __syncthreads();
    {  // GEMM2 (K=512) stats
      const bf16* row = sH1 + m0 * 512;
      v8f acc0 = vzero8(), acc1 = vzero8();
#pragma unroll
      for (int ks = 0; ks < 16; ++ks) {
        v16bf a = ldsA(row, ks * 32, g);
        const bf16* bb = W2p + (size_t)(ks * 16 + w * 2) * 512 + L * 16;  // N/16 = 16
        acc0 = wmma_bf16(a, ldB(bb), acc0);
        acc1 = wmma_bf16(a, ldB(bb + 512), acc1);
      }
      float s0 = 0.f, q0 = 0.f, s1 = 0.f, q1 = 0.f;
#pragma unroll
      for (int r = 0; r < 8; ++r) {
        float z0 = acc0[r] + b2c[0]; s0 += z0; q0 += z0 * z0;
        float z1 = acc1[r] + b2c[1]; s1 += z1; q1 += z1 * z1;
      }
      sumP[0] += s0; sqP[0] += q0; sumP[1] += s1; sqP[1] += q1;
    }
  }
#pragma unroll
  for (int i = 0; i < 2; ++i) {
    float s = sumP[i] + __shfl_xor(sumP[i], 16, 32);
    float q = sqP[i]  + __shfl_xor(sqP[i], 16, 32);
    if (L < 16) {
      int col = w * 32 + i * 16 + L;
      atomicAdd(&sum2[col], s);
      atomicAdd(&sq2[col], q);
    }
  }
}

// ---------------- MLP pass 3: fused full chain -> out ----------------
__global__ __launch_bounds__(256) void mlp_final(
    const float* __restrict__ x2, const int* __restrict__ vi,
    const int* __restrict__ ui, const float* __restrict__ ef,
    const bf16* __restrict__ W1p, const float* __restrict__ b1,
    const float* __restrict__ a1, const float* __restrict__ c1,
    const bf16* __restrict__ W2p, const float* __restrict__ b2,
    const float* __restrict__ a2, const float* __restrict__ c2,
    const bf16* __restrict__ W3p, const float* __restrict__ b3,
    const bf16* __restrict__ W4p, const float* __restrict__ b4,
    const bf16* __restrict__ W5p, const float* __restrict__ b5,
    const float* __restrict__ W6, const float* __restrict__ b6,
    float* __restrict__ out) {
  __shared__ __align__(16) bf16 sPair[16 * 320];
  __shared__ __align__(16) bf16 sH1[16 * 512];
  __shared__ __align__(16) bf16 sH2[16 * 256];
  __shared__ __align__(16) bf16 sH3[16 * 128];
  __shared__ __align__(16) bf16 sH4[16 * 64];
  __shared__ float sH5[16 * 32];
  __shared__ int sV[16], sU[16];
  const int tid = threadIdx.x, w = tid >> 5, L = tid & 31, g = L >> 4, m0 = L & 15;
  const int e0 = blockIdx.x * 16;
  if (tid < 16) { sV[tid] = vi[e0 + tid]; sU[tid] = ui[e0 + tid]; }
  __syncthreads();
  for (int idx = tid; idx < 16 * 320; idx += 256) {
    int m = idx / 320, k = idx - m * 320;
    float v;
    if (k < 128)      v = x2[(size_t)sV[m] * ND + k];
    else if (k < 256) v = x2[(size_t)sU[m] * ND + (k - 128)];
    else              v = ef[(size_t)(e0 + m) * ED + (k - 256)];
    sPair[idx] = (bf16)v;
  }
  __syncthreads();
  {  // GEMM1 + BN1 + relu -> sH1  (each wave: 64 of 512 cols)
    const bf16* row = sPair + m0 * 320;
    v8f acc[4];
#pragma unroll
    for (int i = 0; i < 4; ++i) acc[i] = vzero8();
#pragma unroll
    for (int ks = 0; ks < 10; ++ks) {
      v16bf a = ldsA(row, ks * 32, g);
      const bf16* bb = W1p + (size_t)(ks * 32 + w * 4) * 512 + L * 16;
#pragma unroll
      for (int i = 0; i < 4; ++i) acc[i] = wmma_bf16(a, ldB(bb + i * 512), acc[i]);
    }
#pragma unroll
    for (int i = 0; i < 4; ++i) {
      int col = w * 64 + i * 16 + m0;
      float aa = a1[col];
      float cc = b1[col] * aa + c1[col];
#pragma unroll
      for (int r = 0; r < 8; ++r)
        sH1[(r + 8 * g) * 512 + col] = (bf16)fmaxf(acc[i][r] * aa + cc, 0.0f);
    }
  }
  __syncthreads();
  {  // GEMM2 + BN2 + relu -> sH2  (each wave: 32 of 256 cols)
    const bf16* row = sH1 + m0 * 512;
    v8f acc0 = vzero8(), acc1 = vzero8();
#pragma unroll
    for (int ks = 0; ks < 16; ++ks) {
      v16bf a = ldsA(row, ks * 32, g);
      const bf16* bb = W2p + (size_t)(ks * 16 + w * 2) * 512 + L * 16;
      acc0 = wmma_bf16(a, ldB(bb), acc0);
      acc1 = wmma_bf16(a, ldB(bb + 512), acc1);
    }
    int c0 = w * 32 + m0;
    float aa0 = a2[c0],      cc0 = b2[c0] * aa0 + c2[c0];
    float aa1 = a2[c0 + 16], cc1 = b2[c0 + 16] * aa1 + c2[c0 + 16];
#pragma unroll
    for (int r = 0; r < 8; ++r) {
      int m = r + 8 * g;
      sH2[m * 256 + c0]      = (bf16)fmaxf(acc0[r] * aa0 + cc0, 0.0f);
      sH2[m * 256 + c0 + 16] = (bf16)fmaxf(acc1[r] * aa1 + cc1, 0.0f);
    }
  }
  __syncthreads();
  {  // GEMM3 + relu -> sH3  (each wave: 16 of 128 cols, K=256)
    const bf16* row = sH2 + m0 * 256;
    v8f acc = vzero8();
#pragma unroll
    for (int ks = 0; ks < 8; ++ks) {
      v16bf a = ldsA(row, ks * 32, g);
      const bf16* bb = W3p + (size_t)(ks * 8 + w) * 512 + L * 16;  // N/16 = 8
      acc = wmma_bf16(a, ldB(bb), acc);
    }
    int col = w * 16 + m0;
    float b3c = b3[col];
#pragma unroll
    for (int r = 0; r < 8; ++r)
      sH3[(r + 8 * g) * 128 + col] = (bf16)fmaxf(acc[r] + b3c, 0.0f);
  }
  __syncthreads();
  if (w < 4) {  // GEMM4 + relu -> sH4 (K=128, N=64)
    const bf16* row = sH3 + m0 * 128;
    v8f acc = vzero8();
#pragma unroll
    for (int ks = 0; ks < 4; ++ks) {
      v16bf a = ldsA(row, ks * 32, g);
      const bf16* bb = W4p + (size_t)(ks * 4 + w) * 512 + L * 16;  // N/16 = 4
      acc = wmma_bf16(a, ldB(bb), acc);
    }
    int col = w * 16 + m0;
    float b4c = b4[col];
#pragma unroll
    for (int r = 0; r < 8; ++r)
      sH4[(r + 8 * g) * 64 + col] = (bf16)fmaxf(acc[r] + b4c, 0.0f);
  }
  __syncthreads();
  if (w < 2) {  // GEMM5 + relu -> sH5 (K=64, N=32)
    const bf16* row = sH4 + m0 * 64;
    v8f acc = vzero8();
#pragma unroll
    for (int ks = 0; ks < 2; ++ks) {
      v16bf a = ldsA(row, ks * 32, g);
      const bf16* bb = W5p + (size_t)(ks * 2 + w) * 512 + L * 16;  // N/16 = 2
      acc = wmma_bf16(a, ldB(bb), acc);
    }
    int col = w * 16 + m0;
    float b5c = b5[col];
#pragma unroll
    for (int r = 0; r < 8; ++r)
      sH5[(r + 8 * g) * 32 + col] = fmaxf(acc[r] + b5c, 0.0f);
  }
  __syncthreads();
  if (tid < 16) {  // final 32 -> 1
    float s = b6[0];
#pragma unroll
    for (int k = 0; k < 32; ++k) s += sH5[tid * 32 + k] * W6[k];
    out[e0 + tid] = s;
  }
}

// ---------------- host orchestration ----------------

extern "C" void kernel_launch(void* const* d_in, const int* in_sizes, int n_in,
                              void* d_out, int out_size, void* d_ws, size_t ws_size,
                              hipStream_t stream) {
  (void)n_in; (void)out_size; (void)ws_size;
  const float* x0   = (const float*)d_in[0];
  const int*   eidx = (const int*)d_in[1];
  const float* ef   = (const float*)d_in[2];
  const float* W_nb   = (const float*)d_in[4];
  const float* b_nb   = (const float*)d_in[5];
  const float* W_self = (const float*)d_in[6];
  const float* b_self = (const float*)d_in[7];
  const float* W1 = (const float*)d_in[8];   const float* b1 = (const float*)d_in[9];
  const float* W2 = (const float*)d_in[10];  const float* b2 = (const float*)d_in[11];
  const float* W3 = (const float*)d_in[12];  const float* b3 = (const float*)d_in[13];
  const float* W4 = (const float*)d_in[14];  const float* b4 = (const float*)d_in[15];
  const float* W5 = (const float*)d_in[16];  const float* b5 = (const float*)d_in[17];
  const float* W6 = (const float*)d_in[18];  const float* b6 = (const float*)d_in[19];
  const float* g1 = (const float*)d_in[20];  const float* beta1 = (const float*)d_in[21];
  const float* g2 = (const float*)d_in[22];  const float* beta2 = (const float*)d_in[23];
  float* out = (float*)d_out;

  const int N = in_sizes[0] / ND;   // 50000
  const int E = in_sizes[2] / ED;   // 500000
  const int ET = E / 16;            // 31250 edge tiles
  const int NT = N / 16;            // 3125 node tiles
  const int*  src = eidx;           // edge_index[0]
  const int*  dst = eidx + E;       // edge_index[1]

  // ---- carve workspace (256B aligned slices) ----
  char* wsb = (char*)d_ws;
  size_t off = 0;
  auto alloc = [&](size_t bytes) -> void* {
    void* p = wsb + off;
    off += (bytes + 255) & ~(size_t)255;
    return p;
  };
  bf16* pWnb   = (bf16*)alloc((size_t)192 * 128 * 2);
  bf16* pWself = (bf16*)alloc((size_t)128 * 128 * 2);
  bf16* pW1    = (bf16*)alloc((size_t)320 * 512 * 2);
  bf16* pW2    = (bf16*)alloc((size_t)512 * 256 * 2);
  bf16* pW3    = (bf16*)alloc((size_t)256 * 128 * 2);
  bf16* pW4    = (bf16*)alloc((size_t)128 * 64 * 2);
  bf16* pW5    = (bf16*)alloc((size_t)64 * 32 * 2);
  float* deg   = (float*)alloc((size_t)N * 4);
  float* recip = (float*)alloc((size_t)N * 4);
  float* agg   = (float*)alloc((size_t)N * ND * 4);
  float* x1    = (float*)alloc((size_t)N * ND * 4);
  float* x2    = (float*)alloc((size_t)N * ND * 4);
  float* stats = (float*)alloc((size_t)3072 * 4);
  float* sum1 = stats;        float* sq1 = stats + 512;
  float* sum2 = stats + 1024; float* sq2 = stats + 1280;
  float* a1 = stats + 1536;   float* c1 = stats + 2048;
  float* a2 = stats + 2560;   float* c2 = stats + 2816;

  // ---- pack weights to WMMA fragment layout (bf16) ----
  auto pack = [&](const float* W, bf16* dstp, int K, int Ncols) {
    int total = K * Ncols;
    pack_weight<<<(total + 255) / 256, 256, 0, stream>>>(W, dstp, K, Ncols);
  };
  pack(W_nb, pWnb, 192, 128);
  pack(W_self, pWself, 128, 128);
  pack(W1, pW1, 320, 512);
  pack(W2, pW2, 512, 256);
  pack(W3, pW3, 256, 128);
  pack(W4, pW4, 128, 64);
  pack(W5, pW5, 64, 32);

  // ---- degree / reciprocal ----
  hipMemsetAsync(deg, 0, (size_t)N * 4, stream);
  deg_count<<<(E + 255) / 256, 256, 0, stream>>>(dst, deg, E);
  deg_recip<<<(N + 255) / 256, 256, 0, stream>>>(deg, recip, N);

  // ---- SAGE layer 1: x0 -> x1 ----
  hipMemsetAsync(agg, 0, (size_t)N * ND * 4, stream);
  sage_msg<<<ET, 128, 0, stream>>>(x0, src, dst, ef, pWnb, b_nb, agg);
  sage_node<<<NT, 128, 0, stream>>>(x0, pWself, b_self, agg, recip, x1);

  // ---- SAGE layer 2: x1 -> x2 ----
  hipMemsetAsync(agg, 0, (size_t)N * ND * 4, stream);
  sage_msg<<<ET, 128, 0, stream>>>(x1, src, dst, ef, pWnb, b_nb, agg);
  sage_node<<<NT, 128, 0, stream>>>(x1, pWself, b_self, agg, recip, x2);

  // ---- MLP: batchnorm stats passes + fused final pass ----
  hipMemsetAsync(stats, 0, (size_t)1536 * 4, stream);
  const float invE = 1.0f / (float)E;
  int sblocks = ET < 2048 ? ET : 2048;

  mlp_stats1<<<sblocks, 256, 0, stream>>>(x2, src, dst, ef, pW1, b1, sum1, sq1, ET);
  bn_finalize<<<2, 256, 0, stream>>>(sum1, sq1, g1, beta1, a1, c1, 512, invE);

  mlp_stats2<<<sblocks, 256, 0, stream>>>(x2, src, dst, ef, pW1, b1, a1, c1,
                                          pW2, b2, sum2, sq2, ET);
  bn_finalize<<<1, 256, 0, stream>>>(sum2, sq2, g2, beta2, a2, c2, 256, invE);

  mlp_final<<<ET, 256, 0, stream>>>(x2, src, dst, ef, pW1, b1, a1, c1, pW2, b2,
                                    a2, c2, pW3, b3, pW4, b4, pW5, b5, W6, b6, out);
}
